// GATUpdateAndEdgeTransform_66692252172958
// MI455X (gfx1250) — compile-verified, matrix operand order checked by reading, not run on hardware
//
#include <hip/hip_runtime.h>
#include <hip/hip_bf16.h>
#include <cstdint>
#include <cstddef>

// ---------------------------------------------------------------------------
// GAT (3 layers) for MI455X / gfx1250.
// GEMM h2 = h @ W done with v_wmma_f32_16x16x32_bf16 using a 3-term bf16
// split (hi*hi + hi*lo + lo*hi) to recover ~fp32 accuracy while riding the
// WMMA pipe. Edge softmax/scatter via hardware float atomics.
// ---------------------------------------------------------------------------

typedef __attribute__((ext_vector_type(16))) __bf16 v16bf;
typedef __attribute__((ext_vector_type(8)))  float  v8f;

#define GAT_D 128
#define NEG_SLOPE 0.2f

__device__ __forceinline__ float lrelu(float x) { return x > 0.f ? x : NEG_SLOPE * x; }

// IEEE-float atomic max via integer atomics (monotone under both paths).
__device__ __forceinline__ void atomicMaxF32(float* addr, float val) {
  if (val >= 0.f) atomicMax((int*)addr, __float_as_int(val));
  else            atomicMin((unsigned int*)addr, __float_as_uint(val));
}

__device__ __forceinline__ void atomicAddF32(float* addr, float val) {
  unsafeAtomicAdd(addr, val);   // -> global_atomic_add_f32
}

// --- W prep: split into bf16 hi/lo and transpose (Wt[c][k]) so B-fragment
// loads are contiguous rows. ------------------------------------------------
__global__ void k_prep_w(const float* __restrict__ W,
                         __bf16* __restrict__ wt_hi, __bf16* __restrict__ wt_lo) {
  int idx = blockIdx.x * blockDim.x + threadIdx.x;
  if (idx >= GAT_D * GAT_D) return;
  int k = idx / GAT_D, c = idx % GAT_D;
  float w = W[idx];
  __bf16 hi = (__bf16)w;
  __bf16 lo = (__bf16)(w - (float)hi);
  wt_hi[c * GAT_D + k] = hi;
  wt_lo[c * GAT_D + k] = lo;
}

// --- GEMM: block = 8 waves; block computes rows [16*bid,16*bid+16) x all 128
// cols; wave w owns the 16x16 tile at colBase = 16*w. K-loop 4 x 32, three
// bf16 WMMAs per step (split-precision). A/B fragment layouts follow ISA
// 7.12.2 (16-bit A 16x32; B mirrored; f32 C/D 16x16). -----------------------
__global__ void __launch_bounds__(256)
k_gemm(const float* __restrict__ hin,
       const __bf16* __restrict__ wt_hi, const __bf16* __restrict__ wt_lo,
       float* __restrict__ h2) {
  const int wave    = threadIdx.x >> 5;
  const int lane    = threadIdx.x & 31;
  const int half    = lane >> 4;      // 0: K-low half, 1: K-high half
  const int mrow    = lane & 15;      // A row / B col / D col within tile
  const int row0    = blockIdx.x * 16;
  const int colBase = wave * 16;

  const float*  arow = hin + (size_t)(row0 + mrow) * GAT_D;
  const __bf16* bhiP = wt_hi + (size_t)(colBase + mrow) * GAT_D;
  const __bf16* bloP = wt_lo + (size_t)(colBase + mrow) * GAT_D;

  v8f acc = {};
  #pragma unroll
  for (int ki = 0; ki < GAT_D; ki += 32) {
    const int kbase = ki + 8 * half;
    v16bf ahi, alo, bhi, blo;
    #pragma unroll
    for (int j = 0; j < 8; ++j) {
      float f0 = arow[kbase + j];          // elements 0..7  -> K = kbase+j
      float f1 = arow[kbase + 16 + j];     // elements 8..15 -> K = kbase+16+j
      __bf16 h0 = (__bf16)f0;
      __bf16 h1 = (__bf16)f1;
      ahi[j]     = h0;  alo[j]     = (__bf16)(f0 - (float)h0);
      ahi[8 + j] = h1;  alo[8 + j] = (__bf16)(f1 - (float)h1);
      bhi[j]     = bhiP[kbase + j];
      blo[j]     = bloP[kbase + j];
      bhi[8 + j] = bhiP[kbase + 16 + j];
      blo[8 + j] = bloP[kbase + 16 + j];
    }
    acc = __builtin_amdgcn_wmma_f32_16x16x32_bf16(false, ahi, false, bhi, (short)0, acc, false, false);
    acc = __builtin_amdgcn_wmma_f32_16x16x32_bf16(false, ahi, false, blo, (short)0, acc, false, false);
    acc = __builtin_amdgcn_wmma_f32_16x16x32_bf16(false, alo, false, bhi, (short)0, acc, false, false);
  }
  // D layout: VGPR r, lanes 0-15 -> M=r, lanes 16-31 -> M=r+8; N = lane&15.
  #pragma unroll
  for (int r = 0; r < 8; ++r) {
    int M = r + 8 * half;
    h2[(size_t)(row0 + M) * GAT_D + colBase + mrow] = acc[r];
  }
}

// --- per-node scores s,d + init m=-inf, denom=0 (one wave per row) ---------
__global__ void __launch_bounds__(256)
k_sd_init(const float* __restrict__ h2,
          const float* __restrict__ a_s, const float* __restrict__ a_d,
          float* __restrict__ s, float* __restrict__ d,
          float* __restrict__ m, float* __restrict__ denom, int n) {
  int wave = threadIdx.x >> 5, lane = threadIdx.x & 31;
  int i = blockIdx.x * 8 + wave;
  if (i >= n) return;
  const float* row = h2 + (size_t)i * GAT_D;
  float ps = 0.f, pd = 0.f;
  #pragma unroll
  for (int t = 0; t < 4; ++t) {
    float v = row[lane * 4 + t];
    ps += v * a_s[lane * 4 + t];
    pd += v * a_d[lane * 4 + t];
  }
  #pragma unroll
  for (int off = 16; off >= 1; off >>= 1) {
    ps += __shfl_xor(ps, off, 32);
    pd += __shfl_xor(pd, off, 32);
  }
  if (lane == 0) {
    s[i] = ps; d[i] = pd;
    m[i] = -__builtin_inff();
    denom[i] = 0.f;
  }
}

__global__ void k_zero(float4* __restrict__ p, size_t count4) {
  size_t i = (size_t)blockIdx.x * blockDim.x + threadIdx.x;
  if (i < count4) p[i] = make_float4(0.f, 0.f, 0.f, 0.f);
}

__device__ __forceinline__ void edge_ids(long long tid,
                                         const long long* __restrict__ src,
                                         const long long* __restrict__ dst,
                                         long long E_real, int& sI, int& dI) {
  if (tid < E_real) { sI = (int)src[tid]; dI = (int)dst[tid]; }
  else { int v = (int)(tid - E_real); sI = v; dI = v; }   // self-loop
}

__global__ void k_edge_max(const float* __restrict__ s, const float* __restrict__ d,
                           const long long* __restrict__ src, const long long* __restrict__ dst,
                           long long E_real, long long E_tot, float* __restrict__ m) {
  long long tid = (long long)blockIdx.x * blockDim.x + threadIdx.x;
  if (tid >= E_tot) return;
  int sI, dI; edge_ids(tid, src, dst, E_real, sI, dI);
  atomicMaxF32(&m[dI], lrelu(s[sI] + d[dI]));
}

__global__ void k_edge_sum(const float* __restrict__ s, const float* __restrict__ d,
                           const long long* __restrict__ src, const long long* __restrict__ dst,
                           long long E_real, long long E_tot,
                           const float* __restrict__ m, float* __restrict__ denom) {
  long long tid = (long long)blockIdx.x * blockDim.x + threadIdx.x;
  if (tid >= E_tot) return;
  int sI, dI; edge_ids(tid, src, dst, E_real, sI, dI);
  float e = lrelu(s[sI] + d[dI]);
  atomicAddF32(&denom[dI], expf(e - m[dI]));
}

// one wave per edge; lane handles 4 contiguous features (float4 load).
__global__ void __launch_bounds__(256)
k_edge_scatter(const float* __restrict__ s, const float* __restrict__ d,
               const long long* __restrict__ src, const long long* __restrict__ dst,
               long long E_real, long long E_tot,
               const float* __restrict__ m, const float* __restrict__ denom,
               const float* __restrict__ h2, float* __restrict__ out) {
  int wave = threadIdx.x >> 5, lane = threadIdx.x & 31;
  long long tid = (long long)blockIdx.x * 8 + wave;
  if (tid >= E_tot) return;
  int sI, dI; edge_ids(tid, src, dst, E_real, sI, dI);
  float e = lrelu(s[sI] + d[dI]);
  float alpha = expf(e - m[dI]) / denom[dI];
  const float4 v = ((const float4*)(h2 + (size_t)sI * GAT_D))[lane];
  float* orow = out + (size_t)dI * GAT_D + lane * 4;
  atomicAddF32(orow + 0, alpha * v.x);
  atomicAddF32(orow + 1, alpha * v.y);
  atomicAddF32(orow + 2, alpha * v.z);
  atomicAddF32(orow + 3, alpha * v.w);
}

__global__ void k_finish(const float* __restrict__ accum, const float* __restrict__ bias,
                         float* __restrict__ outp, size_t nD) {
  size_t i = (size_t)blockIdx.x * blockDim.x + threadIdx.x;
  if (i >= nD) return;
  float v = accum[i] + bias[i & (GAT_D - 1)];
  outp[i] = v > 0.f ? v : 0.f;
}

// ---------------------------------------------------------------------------
extern "C" void kernel_launch(void* const* d_in, const int* in_sizes, int n_in,
                              void* d_out, int out_size, void* d_ws, size_t ws_size,
                              hipStream_t stream) {
  const float*     x       = (const float*)d_in[0];
  const long long* ei      = (const long long*)d_in[1];
  const float*     Ws      = (const float*)d_in[2];
  const float*     att_src = (const float*)d_in[3];
  const float*     att_dst = (const float*)d_in[4];
  const float*     bias    = (const float*)d_in[5];

  const int       n      = in_sizes[0] / GAT_D;         // 160000 nodes
  const long long E_real = (long long)in_sizes[1] / 2;  // 1,280,000
  const int       L      = in_sizes[3] / GAT_D;         // 3 layers
  const long long E_tot  = E_real + n;                  // + self loops

  const long long* srcArr = ei;
  const long long* dstArr = ei + E_real;

  // workspace carve-up
  char*  ws     = (char*)d_ws;
  size_t off    = 0;
  float* A      = (float*)(ws + off); off += (size_t)n * GAT_D * sizeof(float);
  float* sArr   = (float*)(ws + off); off += (size_t)n * sizeof(float);
  float* dArr   = (float*)(ws + off); off += (size_t)n * sizeof(float);
  float* mArr   = (float*)(ws + off); off += (size_t)n * sizeof(float);
  float* denArr = (float*)(ws + off); off += (size_t)n * sizeof(float);
  __bf16* wtHi  = (__bf16*)(ws + off); off += (size_t)GAT_D * GAT_D * sizeof(__bf16);
  __bf16* wtLo  = (__bf16*)(ws + off); off += (size_t)GAT_D * GAT_D * sizeof(__bf16);

  float* h2 = (float*)d_out;   // reused as the h2 buffer every layer

  const size_t nD     = (size_t)n * GAT_D;
  const size_t count4 = nD / 4;
  const int    ebNum  = (int)((E_tot + 255) / 256);
  const int    ewNum  = (int)((E_tot + 7) / 8);

  for (int l = 0; l < L; ++l) {
    const float* hin = (l == 0) ? x : A;

    k_prep_w<<<(GAT_D * GAT_D + 255) / 256, 256, 0, stream>>>(
        Ws + (size_t)l * GAT_D * GAT_D, wtHi, wtLo);

    k_gemm<<<n / 16, 256, 0, stream>>>(hin, wtHi, wtLo, h2);

    k_sd_init<<<(n + 7) / 8, 256, 0, stream>>>(
        h2, att_src + (size_t)l * GAT_D, att_dst + (size_t)l * GAT_D,
        sArr, dArr, mArr, denArr, n);

    k_zero<<<(int)((count4 + 255) / 256), 256, 0, stream>>>((float4*)A, count4);

    k_edge_max<<<ebNum, 256, 0, stream>>>(sArr, dArr, srcArr, dstArr, E_real, E_tot, mArr);
    k_edge_sum<<<ebNum, 256, 0, stream>>>(sArr, dArr, srcArr, dstArr, E_real, E_tot, mArr, denArr);
    k_edge_scatter<<<ewNum, 256, 0, stream>>>(sArr, dArr, srcArr, dstArr, E_real, E_tot,
                                              mArr, denArr, h2, A);

    float* fo = (l == L - 1) ? (float*)d_out : A;   // final layer -> d_out
    k_finish<<<(int)((nD + 255) / 256), 256, 0, stream>>>(A, bias + (size_t)l * GAT_D, fo, nD);
  }
}
